// LSTM_Predict_43147241456247
// MI455X (gfx1250) — matrix operand order repaired and checked by reading, not compiled, not run on hardware
//
#include <hip/hip_runtime.h>

// ---------------- problem constants ----------------
#define H        512
#define INDIM    64
#define BATCH    256
#define TSTEPS   512
#define OUTSTEPS 96
#define G4H      2048      // 4*H gate rows
#define BM       16        // batch tile per workgroup (= WMMA M)
#define KT_H     (H/32)    // 16 k-tiles for K=512
#define KT_X     (INDIM/32)// 2 k-tiles for K=64

typedef __attribute__((ext_vector_type(16))) __bf16 bf16x16;
typedef __attribute__((ext_vector_type(8)))  float  v8f;

// ---------------- helpers ----------------
__device__ __forceinline__ unsigned short f2bf(float f) {
  unsigned int u = __float_as_uint(f);
  return (unsigned short)((u + 0x7FFFu + ((u >> 16) & 1u)) >> 16); // RNE
}
__device__ __forceinline__ float sigm(float x)   { return 1.0f / (1.0f + __expf(-x)); }
__device__ __forceinline__ float tanh_f(float x) { float e = __expf(-2.0f * x); return (1.0f - e) / (1.0f + e); }

// Optimization barrier: keeps pointer in SGPRs, stops LICM from hoisting the
// loop-invariant weight stream out of the time loop (round-1 spill disaster).
__device__ __forceinline__ const unsigned short* opq(const unsigned short* p) {
  unsigned long long u = (unsigned long long)p;
  asm volatile("" : "+s"(u));
  return (const unsigned short*)u;
}
__device__ __forceinline__ const float* opqf(const float* p) {
  unsigned long long u = (unsigned long long)p;
  asm volatile("" : "+s"(u));
  return (const float*)u;
}

// Global loads as SGPR64 base + 32-bit element offset: lets the backend pick
// GVS addressing (saddr + voffset + imm) instead of 64-bit VGPR address math.
__device__ __forceinline__ bf16x16 ldg_b(const unsigned short* base, unsigned off) {
  const __attribute__((address_space(1))) unsigned short* gp =
      (const __attribute__((address_space(1))) unsigned short*)(unsigned long long)base;
  return *(const __attribute__((address_space(1))) bf16x16*)(gp + off);
}
__device__ __forceinline__ float ldg_f(const float* base, unsigned off) {
  const __attribute__((address_space(1))) float* gp =
      (const __attribute__((address_space(1))) float*)(unsigned long long)base;
  return gp[off];
}

__device__ __forceinline__ v8f wmma_bf16(bf16x16 a, bf16x16 b, v8f c) {
  return __builtin_amdgcn_wmma_f32_16x16x32_bf16(false, a, false, b, (short)0, c, false, false);
}

// K index inside a 32-wide k-tile for A-matrix VGPR layout (16-bit A 16x32)
__device__ __forceinline__ int a_k(int lane, int j) {
  int v = j >> 1, w = j & 1;
  int k = (v < 4) ? (2 * v + w) : (16 + 2 * (v - 4) + w);
  return k + ((lane >= 16) ? 8 : 0);
}

// ---------------- one LSTM layer for this wave's 64-hidden slice ----------------
template <int KT1, int KT2, bool XW, bool PROJ>
__device__ __forceinline__ void lstm_layer(
    const unsigned short* __restrict__ aBuf1, const unsigned short* W1_,
    const unsigned short* __restrict__ aBuf2, const unsigned short* W2_,
    const float* bias_, const float* xw_, const float* __restrict__ xv,
    float* __restrict__ cLds, unsigned short* __restrict__ hOut,
    const float* pw_, float* __restrict__ yacc,
    int wave, int lane)
{
  const unsigned short* W1 = nullptr;
  if constexpr (KT1 > 0) W1 = opq(W1_);
  const unsigned short* W2 = opq(W2_);
  const float* bias = opqf(bias_);
  const float* xw = nullptr;
  if constexpr (XW) xw = opqf(xw_);
  const float* pw = nullptr;
  if constexpr (PROJ) pw = opqf(pw_);

  const int mbase = (lane >= 16) ? 8 : 0;
  const unsigned nl = (unsigned)(lane & 15);
  const unsigned loff = (unsigned)lane * 16u;

  float part[8];
  if constexpr (PROJ) {
#pragma unroll
    for (int r = 0; r < 8; ++r) part[r] = 0.0f;
  }
  float xvr[8];
  if constexpr (XW) {
#pragma unroll
    for (int r = 0; r < 8; ++r) xvr[r] = xv[r + mbase];
  }

#pragma unroll
  for (int p = 0; p < 2; ++p) {            // two passes of 2 hidden-tiles each
    v8f z[8];                              // [g*2+q] accumulators (64 VGPRs)
#pragma unroll
    for (int g = 0; g < 4; ++g)
#pragma unroll
      for (int q = 0; q < 2; ++q) {
        const unsigned htg = (unsigned)(wave * 4 + p * 2 + q);
        const float bv = ldg_f(bias, (unsigned)g * 512u + htg * 16u + nl);
        v8f zz = {bv, bv, bv, bv, bv, bv, bv, bv};
        z[g * 2 + q] = zz;
      }

    if constexpr (KT1 > 0) {
      const unsigned short* ap = aBuf1 + lane * 16;
#pragma clang loop unroll(disable)
      for (int kt = 0; kt < KT1; ++kt) {
        bf16x16 a = *(const bf16x16*)ap;
        ap += 512;
#pragma unroll
        for (int g = 0; g < 4; ++g)
#pragma unroll
          for (int q = 0; q < 2; ++q) {
            const unsigned htg = (unsigned)(wave * 4 + p * 2 + q);
            const unsigned off = (((unsigned)g * 32u + htg) * (unsigned)KT1 + (unsigned)kt) * 512u + loff;
            z[g * 2 + q] = wmma_bf16(a, ldg_b(W1, off), z[g * 2 + q]);
          }
      }
    }
    {
      const unsigned short* ap = aBuf2 + lane * 16;
#pragma clang loop unroll(disable)
      for (int kt = 0; kt < KT2; ++kt) {
        bf16x16 a = *(const bf16x16*)ap;
        ap += 512;
#pragma unroll
        for (int g = 0; g < 4; ++g)
#pragma unroll
          for (int q = 0; q < 2; ++q) {
            const unsigned htg = (unsigned)(wave * 4 + p * 2 + q);
            const unsigned off = (((unsigned)g * 32u + htg) * (unsigned)KT2 + (unsigned)kt) * 512u + loff;
            z[g * 2 + q] = wmma_bf16(a, ldg_b(W2, off), z[g * 2 + q]);
          }
      }
    }
    if constexpr (XW) {
#pragma unroll
      for (int g = 0; g < 4; ++g)
#pragma unroll
        for (int q = 0; q < 2; ++q) {
          const unsigned htg = (unsigned)(wave * 4 + p * 2 + q);
          const float wv = ldg_f(xw, (unsigned)g * 512u + htg * 16u + nl);
#pragma unroll
          for (int r = 0; r < 8; ++r) z[g * 2 + q][r] += xvr[r] * wv;
        }
    }

    // elementwise cell update + immediate scatter into the *other* h buffer
#pragma unroll
    for (int q = 0; q < 2; ++q) {
      const int htg = wave * 4 + p * 2 + q;
      float* cp = cLds + ((size_t)htg * 32 + lane) * 8;
      v8f c = *(const v8f*)cp;
      float pwv = 0.0f;
      if constexpr (PROJ) pwv = ldg_f(pw, (unsigned)htg * 16u + nl);
      const int k    = htg * 16 + (int)nl;    // hidden index = K of next GEMM
      const int kt2  = k >> 5;
      const int kin  = k & 31;
      const int half = ((kin & 15) >= 8) ? 1 : 0;
      const int jj   = (kin < 16) ? (kin & 7) : (8 + (kin & 7));
#pragma unroll
      for (int r = 0; r < 8; ++r) {
        const float iv = sigm(z[0 * 2 + q][r]);
        const float fv = sigm(z[1 * 2 + q][r]);
        const float gv = tanh_f(z[2 * 2 + q][r]);
        const float ov = sigm(z[3 * 2 + q][r]);
        const float cn = fv * c[r] + iv * gv;
        c[r] = cn;
        const float hn = ov * tanh_f(cn);
        if constexpr (PROJ) part[r] += hn * pwv;
        const int laneA = (r + mbase) + 16 * half;
        hOut[(kt2 * 32 + laneA) * 16 + jj] = f2bf(hn);
      }
      *(v8f*)cp = c;
    }
  }
  if constexpr (PROJ) {
#pragma unroll
    for (int r = 0; r < 8; ++r) atomicAdd(&yacc[r + mbase], part[r]);
  }
}

// ---------------- pre-pass kernels ----------------
__global__ void pack_weight(const float* __restrict__ W, unsigned short* __restrict__ dst, int K) {
  const int KT = K >> 5;
  const size_t total = (size_t)G4H * K;
  size_t idx = (size_t)blockIdx.x * blockDim.x + threadIdx.x;
  if (idx >= total) return;
  const int j    = idx & 15;
  const int lane = (idx >> 4) & 31;
  const size_t rest = idx >> 9;
  const int kt  = (int)(rest % KT);
  const int blk = (int)(rest / KT);
  const int n = (blk >> 5) * 512 + (blk & 31) * 16 + (lane & 15);
  const int k = kt * 32 + ((j & 8) << 1) + ((lane >> 4) << 3) + (j & 7);
  dst[idx] = f2bf(W[(size_t)n * K + k]);
}

__global__ void combine_bias(const float* __restrict__ a, const float* __restrict__ b,
                             float* __restrict__ dst) {
  int i = blockIdx.x * blockDim.x + threadIdx.x;
  if (i < G4H) dst[i] = a[i] + b[i];
}

// ---------------- persistent LSTM: one WG per 16 batch rows ----------------
__global__ __launch_bounds__(256) void lstm_persistent(
    const float* __restrict__ x,
    const unsigned short* __restrict__ PeW0x, const unsigned short* __restrict__ PeW0h,
    const unsigned short* __restrict__ PeW1x, const unsigned short* __restrict__ PeW1h,
    const unsigned short* __restrict__ PdW0h,
    const unsigned short* __restrict__ PdW1x, const unsigned short* __restrict__ PdW1h,
    const float* __restrict__ be0, const float* __restrict__ be1,
    const float* __restrict__ bd0, const float* __restrict__ bd1,
    const float* __restrict__ dWih0, const float* __restrict__ projW,
    const float* __restrict__ projB, float* __restrict__ out)
{
  __shared__ __align__(32) unsigned short xbuf[KT_X * 512];                 // 2 KB
  __shared__ __align__(32) unsigned short h0A[KT_H * 512], h0B[KT_H * 512]; // 2x16 KB
  __shared__ __align__(32) unsigned short h1A[KT_H * 512], h1B[KT_H * 512]; // 2x16 KB
  __shared__ __align__(32) float c0s[32 * 32 * 8];                          // 32 KB
  __shared__ __align__(32) float c1s[32 * 32 * 8];                          // 32 KB
  __shared__ float xdec[BM];
  __shared__ float yacc[BM];

  const int tid  = threadIdx.x;
  const int lane = tid & 31;
  const int wave = tid >> 5;
  const int b0   = blockIdx.x * BM;

  for (int i = tid; i < KT_H * 512; i += 256) { h0A[i] = 0; h1A[i] = 0; }
  for (int i = tid; i < 32 * 32 * 8; i += 256) { c0s[i] = 0.0f; c1s[i] = 0.0f; }

  // gather x(t=0)
#pragma unroll
  for (int s4 = 0; s4 < 4; ++s4) {
    const int s  = tid * 4 + s4;
    const int j  = s & 15;
    const int lg = (s >> 4) & 31;
    const int kt = s >> 9;
    const int m  = lg & 15;
    const int k  = kt * 32 + a_k(lg, j);
    xbuf[s] = f2bf(ldg_f(x, (unsigned)((b0 + m) * TSTEPS + 0) * INDIM + (unsigned)k));
  }
  __syncthreads();

  unsigned short* h0r = h0A; unsigned short* h0w = h0B;
  unsigned short* h1r = h1A; unsigned short* h1w = h1B;

  // ---------------- encoder ----------------
  for (int t = 0; t < TSTEPS; ++t) {
    lstm_layer<KT_X, KT_H, false, false>(xbuf, PeW0x, h0r, PeW0h, be0, nullptr, nullptr,
                                         c0s, h0w, nullptr, nullptr, wave, lane);
    __syncthreads();
    if (t + 1 < TSTEPS) {   // gather next x tile while layer1 runs
#pragma unroll
      for (int s4 = 0; s4 < 4; ++s4) {
        const int s  = tid * 4 + s4;
        const int j  = s & 15;
        const int lg = (s >> 4) & 31;
        const int kt = s >> 9;
        const int m  = lg & 15;
        const int k  = kt * 32 + a_k(lg, j);
        xbuf[s] = f2bf(ldg_f(x, (unsigned)((b0 + m) * TSTEPS + (t + 1)) * INDIM + (unsigned)k));
      }
    }
    lstm_layer<KT_H, KT_H, false, false>(h0w, PeW1x, h1r, PeW1h, be1, nullptr, nullptr,
                                         c1s, h1w, nullptr, nullptr, wave, lane);
    __syncthreads();
    { unsigned short* tp = h0r; h0r = h0w; h0w = tp; }
    { unsigned short* tp = h1r; h1r = h1w; h1w = tp; }
  }

  // ---------------- decoder ----------------
  if (tid < BM) xdec[tid] = 0.0f;
  __syncthreads();
  for (int t = 0; t < OUTSTEPS; ++t) {
    if (tid < BM) yacc[tid] = 0.0f;
    lstm_layer<0, KT_H, true, false>(nullptr, nullptr, h0r, PdW0h, bd0, dWih0, xdec,
                                     c0s, h0w, nullptr, nullptr, wave, lane);
    __syncthreads();
    lstm_layer<KT_H, KT_H, false, true>(h0w, PdW1x, h1r, PdW1h, bd1, nullptr, nullptr,
                                        c1s, h1w, projW, yacc, wave, lane);
    __syncthreads();
    if (tid < BM) {
      const float y = yacc[tid] + ldg_f(projB, 0u);
      xdec[tid] = y;
      out[(size_t)(b0 + tid) * OUTSTEPS + t] = y;
    }
    __syncthreads();
    { unsigned short* tp = h0r; h0r = h0w; h0w = tp; }
    { unsigned short* tp = h1r; h1r = h1w; h1w = tp; }
  }
}

// ---------------- launch ----------------
extern "C" void kernel_launch(void* const* d_in, const int* in_sizes, int n_in,
                              void* d_out, int out_size, void* d_ws, size_t ws_size,
                              hipStream_t stream) {
  (void)in_sizes; (void)n_in; (void)out_size; (void)ws_size;
  const float* x     = (const float*)d_in[0];
  const float* eWih0 = (const float*)d_in[1];
  const float* eWhh0 = (const float*)d_in[2];
  const float* ebih0 = (const float*)d_in[3];
  const float* ebhh0 = (const float*)d_in[4];
  const float* eWih1 = (const float*)d_in[5];
  const float* eWhh1 = (const float*)d_in[6];
  const float* ebih1 = (const float*)d_in[7];
  const float* ebhh1 = (const float*)d_in[8];
  const float* dWih0 = (const float*)d_in[9];
  const float* dWhh0 = (const float*)d_in[10];
  const float* dbih0 = (const float*)d_in[11];
  const float* dbhh0 = (const float*)d_in[12];
  const float* dWih1 = (const float*)d_in[13];
  const float* dWhh1 = (const float*)d_in[14];
  const float* dbih1 = (const float*)d_in[15];
  const float* dbhh1 = (const float*)d_in[16];
  const float* projW = (const float*)d_in[17];
  const float* projB = (const float*)d_in[18];
  float* out = (float*)d_out;

  unsigned short* ws = (unsigned short*)d_ws;
  size_t o = 0;
  unsigned short* P_eW0x = ws + o; o += (size_t)G4H * INDIM;
  unsigned short* P_eW0h = ws + o; o += (size_t)G4H * H;
  unsigned short* P_eW1x = ws + o; o += (size_t)G4H * H;
  unsigned short* P_eW1h = ws + o; o += (size_t)G4H * H;
  unsigned short* P_dW0h = ws + o; o += (size_t)G4H * H;
  unsigned short* P_dW1x = ws + o; o += (size_t)G4H * H;
  unsigned short* P_dW1h = ws + o; o += (size_t)G4H * H;
  o = (o + 1) & ~(size_t)1;
  float* be0 = (float*)(ws + o); o += 2 * G4H;
  float* be1 = (float*)(ws + o); o += 2 * G4H;
  float* bd0 = (float*)(ws + o); o += 2 * G4H;
  float* bd1 = (float*)(ws + o); o += 2 * G4H;

  const dim3 blk(256);
  pack_weight<<<(G4H * INDIM + 255) / 256, blk, 0, stream>>>(eWih0, P_eW0x, INDIM);
  pack_weight<<<(G4H * H + 255) / 256, blk, 0, stream>>>(eWhh0, P_eW0h, H);
  pack_weight<<<(G4H * H + 255) / 256, blk, 0, stream>>>(eWih1, P_eW1x, H);
  pack_weight<<<(G4H * H + 255) / 256, blk, 0, stream>>>(eWhh1, P_eW1h, H);
  pack_weight<<<(G4H * H + 255) / 256, blk, 0, stream>>>(dWhh0, P_dW0h, H);
  pack_weight<<<(G4H * H + 255) / 256, blk, 0, stream>>>(dWih1, P_dW1x, H);
  pack_weight<<<(G4H * H + 255) / 256, blk, 0, stream>>>(dWhh1, P_dW1h, H);
  combine_bias<<<G4H / 256, blk, 0, stream>>>(ebih0, ebhh0, be0);
  combine_bias<<<G4H / 256, blk, 0, stream>>>(ebih1, ebhh1, be1);
  combine_bias<<<G4H / 256, blk, 0, stream>>>(dbih0, dbhh0, bd0);
  combine_bias<<<G4H / 256, blk, 0, stream>>>(dbih1, dbhh1, bd1);

  lstm_persistent<<<BATCH / BM, blk, 0, stream>>>(
      x, P_eW0x, P_eW0h, P_eW1x, P_eW1h, P_dW0h, P_dW1x, P_dW1h,
      be0, be1, bd0, bd1, dWih0, projW, projB, out);
}
